// EncoderLayer_47399259079139
// MI455X (gfx1250) — compile-verified, matrix operand order checked by reading, not run on hardware
//
#include <hip/hip_runtime.h>
#include <hip/hip_bf16.h>

// ---------------------------------------------------------------------------
// Transformer encoder layer for MI455X (gfx1250, wave32, WMMA + TDM).
// B=2, S=2048, D=1024, H=16, dk=64, DFF=2048.  All GEMMs via
// v_wmma_f32_16x16x32_f16; GEMM tiles staged into LDS by the Tensor Data
// Mover (tensor_load_to_lds, TENSORcnt) with double buffering.
// ---------------------------------------------------------------------------

#define S_LEN   2048
#define D_MOD   1024
#define N_HEAD  16
#define D_HEAD  64
#define D_FF    2048
#define BATCH   2
#define BS      (BATCH * S_LEN)      // 4096 rows
#define LN_EPS  1e-6f

typedef __attribute__((ext_vector_type(16))) _Float16 v16h;
typedef __attribute__((ext_vector_type(8)))  _Float16 v8h;
typedef __attribute__((ext_vector_type(8)))  float    v8f;
typedef __attribute__((ext_vector_type(4)))  unsigned int v4u;
typedef __attribute__((ext_vector_type(8)))  int      v8i_t;
typedef __attribute__((ext_vector_type(4)))  int      v4i_t;

#if defined(__has_builtin)
#if __has_builtin(__builtin_amdgcn_tensor_load_to_lds) && \
    __has_builtin(__builtin_amdgcn_s_wait_tensorcnt)
#define USE_TDM 1
#endif
#endif
#ifndef USE_TDM
#define USE_TDM 0
#endif

// Load a WMMA f16 operand fragment (A layout == B layout, 16x32, per ISA
// 7.12.2): element e<8  -> K = e + laneHalf*8          (chunk 0, 16 bytes)
//          element e>=8 -> K = 16 + (e-8) + laneHalf*8 (chunk 1, 16 bytes)
static __device__ __forceinline__ v16h load_frag(const _Float16* base, int laneHalf) {
  v16h r;
  *((v8h*)&r)       = *(const v8h*)(base + laneHalf * 8);
  *(((v8h*)&r) + 1) = *(const v8h*)(base + 16 + laneHalf * 8);
  return r;
}

static __device__ __forceinline__ v8f wmma_f16(v16h a, v16h b, v8f c) {
  return __builtin_amdgcn_wmma_f32_16x16x32_f16(
      /*neg_a=*/false, a, /*neg_b=*/false, b,
      /*c_mod=*/(short)0, c, /*reuse_a=*/false, /*reuse_b=*/false);
}

static __device__ __forceinline__ float gelu_exact(float x) {
  return 0.5f * x * (1.0f + erff(x * 0.70710678118654752f));
}

#if USE_TDM
// Issue a 2D TDM load: tile (tile1 rows x tile0 f16 elements) from a
// row-major f16 tensor with row stride `stride0` (elements) into LDS at byte
// offset `lds_off`.  D# bit packing per CDNA5 ISA:
//   group0: [1:0]=count=1, [63:32]=lds_addr, [120:64]=global_addr, [127:126]=2
//   group1: [17:16]=data_size(1 => 2B), [79:48]=tensor_dim0, [111:80]=dim1,
//           [127:112]=tile_dim0, [143:128]=tile_dim1, [207:160]=dim0_stride
// This toolchain exposes the 6-arg builtin:
//   (uint32x4 g0, int32x8 g1, int32x4, int32x4, int32x8, i32 cpol)
static __device__ __forceinline__ void tdm_load_2d(const void* gsrc,
                                                   unsigned lds_off,
                                                   unsigned dim0, unsigned dim1,
                                                   unsigned tile0, unsigned tile1,
                                                   unsigned long long stride0) {
  unsigned long long ga = (unsigned long long)gsrc;
  v4u g0;
  g0[0] = 1u;                                   // count=1, user descriptor
  g0[1] = lds_off;                              // lds_addr
  g0[2] = (unsigned)ga;                         // global_addr[31:0]
  g0[3] = (unsigned)((ga >> 32) & 0x1FFFFFFu) | (2u << 30);  // addr[56:32]|type
  v8i_t g1;
  g1[0] = (int)(1u << 16);                      // data_size=1 (2 bytes)
  g1[1] = (int)((dim0 & 0xFFFFu) << 16);        // tensor_dim0[15:0]
  g1[2] = (int)((dim0 >> 16) | ((dim1 & 0xFFFFu) << 16));
  g1[3] = (int)((dim1 >> 16) | (tile0 << 16));  // tile_dim0
  g1[4] = (int)tile1;                           // tile_dim1 (tile_dim2=0)
  g1[5] = (int)(unsigned)(stride0 & 0xFFFFFFFFull);
  g1[6] = (int)(unsigned)((stride0 >> 32) & 0xFFFFull);  // dim1_stride=0
  g1[7] = 0;
  v4i_t z4 = {0, 0, 0, 0};
  v8i_t z8 = {0, 0, 0, 0, 0, 0, 0, 0};
  __builtin_amdgcn_tensor_load_to_lds(g0, g1, z4, z4, z8, 0);
}
#endif

// ---------------------------------------------------------------------------
// LayerNorm (torch-style: alpha*(x-mean)/(std+eps)+bias, std unbiased ddof=1)
// ---------------------------------------------------------------------------
__global__ __launch_bounds__(256) void ln_kernel(const float* __restrict__ x,
                                                 const float* __restrict__ alpha,
                                                 const float* __restrict__ bias,
                                                 _Float16* __restrict__ out) {
  const int row = blockIdx.x;
  const int t = threadIdx.x;
  const float4* xr = (const float4*)(x + (size_t)row * D_MOD);
  float4 f = xr[t];

  __shared__ float ssum[256];
  __shared__ float ssq[256];
  ssum[t] = f.x + f.y + f.z + f.w;
  ssq[t]  = f.x * f.x + f.y * f.y + f.z * f.z + f.w * f.w;
  __syncthreads();
  for (int off = 128; off > 0; off >>= 1) {
    if (t < off) { ssum[t] += ssum[t + off]; ssq[t] += ssq[t + off]; }
    __syncthreads();
  }
  const float mean = ssum[0] * (1.0f / D_MOD);
  const float var  = (ssq[0] - (float)D_MOD * mean * mean) * (1.0f / (D_MOD - 1));
  const float inv  = 1.0f / (sqrtf(fmaxf(var, 0.0f)) + LN_EPS);

  const int c = t * 4;
  _Float16* orow = out + (size_t)row * D_MOD;
  orow[c + 0] = (_Float16)(alpha[c + 0] * (f.x - mean) * inv + bias[c + 0]);
  orow[c + 1] = (_Float16)(alpha[c + 1] * (f.y - mean) * inv + bias[c + 1]);
  orow[c + 2] = (_Float16)(alpha[c + 2] * (f.z - mean) * inv + bias[c + 2]);
  orow[c + 3] = (_Float16)(alpha[c + 3] * (f.w - mean) * inv + bias[c + 3]);
}

// ---------------------------------------------------------------------------
// Weight transpose + f32->f16 convert:  Wt[n,k] = (f16) W[k,n].
// ---------------------------------------------------------------------------
__global__ __launch_bounds__(256) void wtrans_kernel(const float* __restrict__ W,
                                                     _Float16* __restrict__ Wt,
                                                     int K, int N) {
  __shared__ _Float16 tile[16][17];
  const int n0 = blockIdx.x * 16;
  const int k0 = blockIdx.y * 16;
  tile[threadIdx.y][threadIdx.x] =
      (_Float16)W[(size_t)(k0 + threadIdx.y) * N + (n0 + threadIdx.x)];
  __syncthreads();
  Wt[(size_t)(n0 + threadIdx.y) * K + (k0 + threadIdx.x)] = tile[threadIdx.x][threadIdx.y];
}

// ---------------------------------------------------------------------------
// WMMA GEMM with TDM-staged, double-buffered LDS tiles.
//   C[M,N] = A[M,K] x Bt[N,K]^T + bias[N] (+GELU) (+f32 residual)
// Block tile 128x128, K-step 32.  Wave w owns rows [w*16, w*16+16) x 128 cols
// (8 accumulators, 8 WMMAs per K-step).  Wave 0 drives the TDM: issues next
// buffer's loads, computes, s_wait_tensorcnt, then the workgroup barrier
// publishes the staged tile.  grid = (N/128, M/128), block = 256.
// ---------------------------------------------------------------------------
__global__ __launch_bounds__(256) void gemm_wmma_kernel(
    const _Float16* __restrict__ A, const _Float16* __restrict__ Bt,
    const float* __restrict__ bias, const float* __restrict__ residual,
    float* __restrict__ outF, _Float16* __restrict__ outH,
    int M, int N, int K, int act, int headT) {
  __shared__ _Float16 ldsA[2][128 * 32];   // 16 KB
  __shared__ _Float16 ldsB[2][128 * 32];   // 16 KB

  const int lane = threadIdx.x & 31;
  const int wv   = threadIdx.x >> 5;
  const int half = lane >> 4;
  const int l16  = lane & 15;
  const int mBlock = blockIdx.y * 128;
  const int nBlock = blockIdx.x * 128;

  const _Float16* aG = A + (size_t)mBlock * K;    // 128 rows, stride K
  const _Float16* bG = Bt + (size_t)nBlock * K;   // 128 rows, stride K

  v8f acc[8] = {};

#if USE_TDM
  const unsigned aOff[2] = {(unsigned)(uintptr_t)&ldsA[0][0],
                            (unsigned)(uintptr_t)&ldsA[1][0]};
  const unsigned bOff[2] = {(unsigned)(uintptr_t)&ldsB[0][0],
                            (unsigned)(uintptr_t)&ldsB[1][0]};
  if (wv == 0) {
    tdm_load_2d(aG, aOff[0], (unsigned)K, 128u, 32u, 128u, (unsigned long long)K);
    tdm_load_2d(bG, bOff[0], (unsigned)K, 128u, 32u, 128u, (unsigned long long)K);
    __builtin_amdgcn_s_wait_tensorcnt(0);
  }
  __syncthreads();
#else
  // Fallback: cooperative staging (256 threads; thread t -> row t/2, chunk t%2)
  {
    const int tr = threadIdx.x >> 1, tc = (threadIdx.x & 1) * 16;
    const v8h* sa = (const v8h*)(aG + (size_t)tr * K + tc);
    const v8h* sb = (const v8h*)(bG + (size_t)tr * K + tc);
    v8h* da = (v8h*)(&ldsA[0][tr * 32 + tc]);
    v8h* db = (v8h*)(&ldsB[0][tr * 32 + tc]);
    da[0] = sa[0]; da[1] = sa[1];
    db[0] = sb[0]; db[1] = sb[1];
  }
  __syncthreads();
#endif

  int buf = 0;
  for (int k0 = 0; k0 < K; k0 += 32) {
    const bool more = (k0 + 32) < K;
#if USE_TDM
    if (wv == 0 && more) {
      tdm_load_2d(aG + k0 + 32, aOff[buf ^ 1], (unsigned)K, 128u, 32u, 128u,
                  (unsigned long long)K);
      tdm_load_2d(bG + k0 + 32, bOff[buf ^ 1], (unsigned)K, 128u, 32u, 128u,
                  (unsigned long long)K);
    }
#else
    if (more) {
      const int tr = threadIdx.x >> 1, tc = (threadIdx.x & 1) * 16;
      const v8h* sa = (const v8h*)(aG + (size_t)tr * K + k0 + 32 + tc);
      const v8h* sb = (const v8h*)(bG + (size_t)tr * K + k0 + 32 + tc);
      v8h* da = (v8h*)(&ldsA[buf ^ 1][tr * 32 + tc]);
      v8h* db = (v8h*)(&ldsB[buf ^ 1][tr * 32 + tc]);
      da[0] = sa[0]; da[1] = sa[1];
      db[0] = sb[0]; db[1] = sb[1];
    }
#endif
    // compute from current buffer
    const v16h a = load_frag(&ldsA[buf][(wv * 16 + l16) * 32], half);
#pragma unroll
    for (int tI = 0; tI < 8; tI++) {
      const v16h b = load_frag(&ldsB[buf][(tI * 16 + l16) * 32], half);
      acc[tI] = wmma_f16(a, b, acc[tI]);
    }
#if USE_TDM
    if (wv == 0 && more) __builtin_amdgcn_s_wait_tensorcnt(0);
#endif
    __syncthreads();
    buf ^= 1;
  }

  // epilogue
#pragma unroll
  for (int tI = 0; tI < 8; tI++) {
    const int n = nBlock + tI * 16 + l16;
    const float bn = bias ? bias[n] : 0.0f;
#pragma unroll
    for (int r = 0; r < 8; r++) {
      const int m = mBlock + wv * 16 + r + half * 8;
      float v = acc[tI][r] + bn;
      if (act == 1) v = gelu_exact(v);
      if (residual) v += residual[(size_t)m * N + n];
      if (outF) outF[(size_t)m * N + n] = v;
      if (outH) {
        if (headT) {
          // V^T layout: vt[b][h][d][s], m = b*S + s, n = h*64 + d
          const int b = m >> 11, s = m & (S_LEN - 1);
          const int h = n >> 6, d = n & (D_HEAD - 1);
          outH[(((size_t)(b * N_HEAD + h) * D_HEAD + d) << 11) + s] = (_Float16)v;
        } else {
          outH[(size_t)m * N + n] = (_Float16)v;
        }
      }
    }
  }
}

// ---------------------------------------------------------------------------
// Flash attention (no S x S materialization).  One wave owns 16 query rows of
// one (b,h); block = 8 waves = 128 queries.  grid = (B*H, S/128).
// ---------------------------------------------------------------------------
__global__ __launch_bounds__(256) void attn_kernel(
    const _Float16* __restrict__ q, const _Float16* __restrict__ k,
    const _Float16* __restrict__ vt, const int* __restrict__ mask,
    _Float16* __restrict__ out) {
  __shared__ _Float16 ldsP[8][16 * 32];  // per-wave P staging (1 KB each)

  const int lane = threadIdx.x & 31;
  const int wv   = threadIdx.x >> 5;
  const int half = lane >> 4;
  const int l16  = lane & 15;
  const int bh = blockIdx.x;
  const int b = bh >> 4;
  const int h = bh & 15;
  const int q0 = blockIdx.y * 128 + wv * 16;

  const _Float16* qrow = q + (size_t)(b * S_LEN + q0 + l16) * D_MOD + h * D_HEAD;
  const v16h qa0 = load_frag(qrow, half);        // dk 0..31
  const v16h qa1 = load_frag(qrow + 32, half);   // dk 32..63

  const _Float16* kbase = k + (size_t)(b * S_LEN) * D_MOD + h * D_HEAD;
  const _Float16* vbase = vt + (size_t)(b * N_HEAD + h) * D_HEAD * S_LEN;  // [64][2048]
  const int* maskb = mask + (size_t)b * S_LEN * S_LEN;
  _Float16* myP = &ldsP[wv][0];

  float mrow[8], lrow[8];
  v8f acc0 = {}, acc1 = {}, acc2 = {}, acc3 = {};
#pragma unroll
  for (int r = 0; r < 8; r++) { mrow[r] = -3.0e38f; lrow[r] = 0.0f; }

  for (int j0 = 0; j0 < S_LEN; j0 += 32) {
    // ---- scores: S = (Q K^T) / sqrt(dk), two 16-key tiles -----------------
    const _Float16* kr0 = kbase + (size_t)(j0 + l16) * D_MOD;
    const _Float16* kr1 = kbase + (size_t)(j0 + 16 + l16) * D_MOD;
    v8f s0 = {}, s1 = {};
    s0 = wmma_f16(qa0, load_frag(kr0, half), s0);
    s0 = wmma_f16(qa1, load_frag(kr0 + 32, half), s0);
    s1 = wmma_f16(qa0, load_frag(kr1, half), s1);
    s1 = wmma_f16(qa1, load_frag(kr1 + 32, half), s1);

    // ---- online softmax ---------------------------------------------------
#pragma unroll
    for (int r = 0; r < 8; r++) {
      const int m = q0 + r + half * 8;
      const int* mr = maskb + (size_t)m * S_LEN + j0;
      float a0 = s0[r] * 0.125f;  // 1/sqrt(64)
      float a1 = s1[r] * 0.125f;
      if (mr[l16] == 0)      a0 = -1.0e9f;
      if (mr[16 + l16] == 0) a1 = -1.0e9f;

      float loc = fmaxf(a0, a1);
      loc = fmaxf(loc, __shfl_xor(loc, 8, 16));
      loc = fmaxf(loc, __shfl_xor(loc, 4, 16));
      loc = fmaxf(loc, __shfl_xor(loc, 2, 16));
      loc = fmaxf(loc, __shfl_xor(loc, 1, 16));
      const float mn = fmaxf(mrow[r], loc);
      const float sc = __expf(mrow[r] - mn);
      const float p0 = __expf(a0 - mn);
      const float p1 = __expf(a1 - mn);
      float rs = p0 + p1;
      rs += __shfl_xor(rs, 8, 16);
      rs += __shfl_xor(rs, 4, 16);
      rs += __shfl_xor(rs, 2, 16);
      rs += __shfl_xor(rs, 1, 16);
      lrow[r] = lrow[r] * sc + rs;
      mrow[r] = mn;
      acc0[r] *= sc; acc1[r] *= sc; acc2[r] *= sc; acc3[r] *= sc;

      const int mm = r + half * 8;  // row within tile
      myP[mm * 32 + l16]      = (_Float16)p0;
      myP[mm * 32 + 16 + l16] = (_Float16)p1;
    }
    asm volatile("s_wait_dscnt 0" ::: "memory");  // cross-lane LDS RAW

    // ---- P (A operand, 16x32) and P*V ------------------------------------
    const v16h pa = load_frag(myP + l16 * 32, half);
    {
      const _Float16* vr0 = vbase + (size_t)(0 * 16 + l16) * S_LEN + j0;
      const _Float16* vr1 = vbase + (size_t)(1 * 16 + l16) * S_LEN + j0;
      const _Float16* vr2 = vbase + (size_t)(2 * 16 + l16) * S_LEN + j0;
      const _Float16* vr3 = vbase + (size_t)(3 * 16 + l16) * S_LEN + j0;
      acc0 = wmma_f16(pa, load_frag(vr0, half), acc0);
      acc1 = wmma_f16(pa, load_frag(vr1, half), acc1);
      acc2 = wmma_f16(pa, load_frag(vr2, half), acc2);
      acc3 = wmma_f16(pa, load_frag(vr3, half), acc3);
    }
  }

  // ---- normalize + write f16 [BS, D] --------------------------------------
  v8f accs[4] = {acc0, acc1, acc2, acc3};
#pragma unroll
  for (int r = 0; r < 8; r++) {
    const int m = q0 + r + half * 8;
    const float inv = 1.0f / lrow[r];
    _Float16* orow = out + (size_t)(b * S_LEN + m) * D_MOD + h * D_HEAD;
#pragma unroll
    for (int tI = 0; tI < 4; tI++)
      orow[tI * 16 + l16] = (_Float16)(accs[tI][r] * inv);
  }
}

// ---------------------------------------------------------------------------
// Host-side orchestration.
// ---------------------------------------------------------------------------
extern "C" void kernel_launch(void* const* d_in, const int* in_sizes, int n_in,
                              void* d_out, int out_size, void* d_ws, size_t ws_size,
                              hipStream_t stream) {
  (void)in_sizes; (void)n_in; (void)out_size; (void)ws_size;
  const float* x      = (const float*)d_in[0];
  const int*   mask   = (const int*)d_in[1];
  const float* alpha1 = (const float*)d_in[2];
  const float* bias1  = (const float*)d_in[3];
  const float* Wq = (const float*)d_in[4];  const float* bq = (const float*)d_in[5];
  const float* Wk = (const float*)d_in[6];  const float* bk = (const float*)d_in[7];
  const float* Wv = (const float*)d_in[8];  const float* bv = (const float*)d_in[9];
  const float* Wo = (const float*)d_in[10]; const float* bo = (const float*)d_in[11];
  const float* alpha2 = (const float*)d_in[12];
  const float* bias2  = (const float*)d_in[13];
  const float* W1 = (const float*)d_in[14]; const float* b1 = (const float*)d_in[15];
  const float* W2 = (const float*)d_in[16]; const float* b2 = (const float*)d_in[17];
  float* out = (float*)d_out;

  // ---- workspace carve (f16 unless noted); ~96 MB total --------------------
  char* w = (char*)d_ws;
  auto carveH = [&](size_t n) { _Float16* p = (_Float16*)w; w += n * sizeof(_Float16); return p; };
  auto carveF = [&](size_t n) { float* p = (float*)w; w += n * sizeof(float); return p; };
  _Float16* x2f   = carveH((size_t)BS * D_MOD);        // LN1 output
  _Float16* Wqt   = carveH((size_t)D_MOD * D_MOD);
  _Float16* Wkt   = carveH((size_t)D_MOD * D_MOD);
  _Float16* Wvt   = carveH((size_t)D_MOD * D_MOD);
  _Float16* Wot   = carveH((size_t)D_MOD * D_MOD);
  _Float16* W1t   = carveH((size_t)D_FF * D_MOD);      // [DFF, D]
  _Float16* W2t   = carveH((size_t)D_MOD * D_FF);      // [D, DFF]
  _Float16* qf    = carveH((size_t)BS * D_MOD);
  _Float16* kf    = carveH((size_t)BS * D_MOD);
  _Float16* vtf   = carveH((size_t)BS * D_MOD);        // V^T [B,H,dk,S]
  _Float16* att   = carveH((size_t)BS * D_MOD);
  float*    xnew  = carveF((size_t)BS * D_MOD);        // x + attn (f32)
  _Float16* y2f   = carveH((size_t)BS * D_MOD);        // LN2 output
  _Float16* h1f   = carveH((size_t)BS * D_FF);         // gelu(ffn1)

  const dim3 blk256(256), blkT(16, 16);

  // 1) LN1
  ln_kernel<<<BS, blk256, 0, stream>>>(x, alpha1, bias1, x2f);

  // 2) transpose+convert weights
  wtrans_kernel<<<dim3(D_MOD / 16, D_MOD / 16), blkT, 0, stream>>>(Wq, Wqt, D_MOD, D_MOD);
  wtrans_kernel<<<dim3(D_MOD / 16, D_MOD / 16), blkT, 0, stream>>>(Wk, Wkt, D_MOD, D_MOD);
  wtrans_kernel<<<dim3(D_MOD / 16, D_MOD / 16), blkT, 0, stream>>>(Wv, Wvt, D_MOD, D_MOD);
  wtrans_kernel<<<dim3(D_MOD / 16, D_MOD / 16), blkT, 0, stream>>>(Wo, Wot, D_MOD, D_MOD);
  wtrans_kernel<<<dim3(D_FF / 16, D_MOD / 16), blkT, 0, stream>>>(W1, W1t, D_MOD, D_FF);
  wtrans_kernel<<<dim3(D_MOD / 16, D_FF / 16), blkT, 0, stream>>>(W2, W2t, D_FF, D_MOD);

  // 3) QKV projections (V stored head-transposed)
  const dim3 gQ(D_MOD / 128, BS / 128);
  gemm_wmma_kernel<<<gQ, blk256, 0, stream>>>(x2f, Wqt, bq, nullptr, nullptr, qf,
                                              BS, D_MOD, D_MOD, 0, 0);
  gemm_wmma_kernel<<<gQ, blk256, 0, stream>>>(x2f, Wkt, bk, nullptr, nullptr, kf,
                                              BS, D_MOD, D_MOD, 0, 0);
  gemm_wmma_kernel<<<gQ, blk256, 0, stream>>>(x2f, Wvt, bv, nullptr, nullptr, vtf,
                                              BS, D_MOD, D_MOD, 0, 1);

  // 4) flash attention
  attn_kernel<<<dim3(BATCH * N_HEAD, S_LEN / 128), blk256, 0, stream>>>(qf, kf, vtf, mask, att);

  // 5) output projection + residual -> xnew (f32)
  gemm_wmma_kernel<<<gQ, blk256, 0, stream>>>(att, Wot, bo, x, xnew, nullptr,
                                              BS, D_MOD, D_MOD, 0, 0);

  // 6) LN2
  ln_kernel<<<BS, blk256, 0, stream>>>(xnew, alpha2, bias2, y2f);

  // 7) FFN1: gelu(y2 @ W1 + b1) -> h1 (f16)
  gemm_wmma_kernel<<<dim3(D_FF / 128, BS / 128), blk256, 0, stream>>>(
      y2f, W1t, b1, nullptr, nullptr, h1f, BS, D_FF, D_MOD, 1, 0);

  // 8) FFN2: out = xnew + gelu(h1 @ W2 + b2)  (f32 -> d_out)
  gemm_wmma_kernel<<<dim3(D_MOD / 128, BS / 128), blk256, 0, stream>>>(
      h1f, W2t, b2, xnew, out, nullptr, BS, D_MOD, D_FF, 1, 0);
}
